// TransformerConv_Encoder_50165218018090
// MI455X (gfx1250) — matrix operand order repaired and checked by reading, not compiled, not run on hardware
//
#include <hip/hip_runtime.h>

// ---------------------------------------------------------------------------
// TransformerConv x2 for MI455X (gfx1250, wave32).
//  - Dense projections via v_wmma_f32_16x16x32_f16 (fp32 accumulate),
//    weights pre-packed to f16 in WMMA B-fragment order (32B vector loads),
//    A fragments staged through LDS in fragment order (32B ds loads).
//  - Edge phase: gather from the L2-resident P=[Q|K|V|S] table (90MB << 192MB
//    L2), segment softmax via L2 atomics, head-mean fused into the scatter.
// ---------------------------------------------------------------------------

typedef __attribute__((ext_vector_type(16))) _Float16 v16h;
typedef __attribute__((ext_vector_type(4)))  _Float16 v4h;
typedef __attribute__((ext_vector_type(8)))  float    v8f;

#define NNODES 50000
#define NEDGES 800000
#define DIN    64
#define NH     2
#define CH     64
#define DQK    128            // NH * CH
#define PW     448            // per-node row: Q(128) K(128) V(128) S(64)
#define NTILES 28             // 448 / 16
#define BPACKN (NTILES * 2 * 32 * 16)   // packed B elements (halves)

__device__ __forceinline__ float atomicMaxF(float* addr, float v) {
  // addr initialized to -inf; float order == int order for >=0, reversed uint
  // order for negatives.
  if (v >= 0.0f) {
    return __int_as_float(atomicMax((int*)addr, __float_as_int(v)));
  } else {
    return __uint_as_float(atomicMin((unsigned int*)addr, __float_as_uint(v)));
  }
}

// ---------------------------------------------------------------------------
// One-off per layer: convert fp32 weights to f16 in a contiguous-k (A-matching) WMMA B-fragment
// layout, and concatenate the four bias vectors.
//   bpack[((nt*2+s)*32 + lane)*16 + e] = W_grp[(32*s + (lane<16? e : e+16))*dout
//                                              + (nt&7)*16 + (lane&15)]
// ---------------------------------------------------------------------------
__global__ void pack_weights(const float* __restrict__ wq, const float* __restrict__ bq,
                             const float* __restrict__ wk, const float* __restrict__ bk,
                             const float* __restrict__ wv, const float* __restrict__ bv,
                             const float* __restrict__ wsk, const float* __restrict__ bsk,
                             _Float16* __restrict__ bpack, float* __restrict__ bcat)
{
  int idx = blockIdx.x * 256 + threadIdx.x;
  const float* Wt[4] = {wq, wk, wv, wsk};
  const float* Bt[4] = {bq, bk, bv, bsk};
  if (idx < BPACKN) {
    int e    = idx & 15;
    int lane = (idx >> 4) & 31;
    int s    = (idx >> 9) & 1;
    int nt   = idx >> 10;
    int grp  = nt >> 3;
    int dout = (grp == 3) ? CH : DQK;
    int n    = ((nt & 7) << 4) + (lane & 15);
    int K    = 32 * s + ((lane < 16) ? e : (e + 16));
    bpack[idx] = (_Float16)Wt[grp][K * dout + n];
  }
  if (idx < PW) {
    int grp = idx >> 7;                       // 0..3 (q,k,v,skip)
    bcat[idx] = Bt[grp][idx - grp * 128];
  }
}

// ---------------------------------------------------------------------------
// Fused Q/K/V/skip projection: [16 x 64] x-tile @ [64 x 448] packed weights.
// 28 n-tiles over 8 waves; 2 WMMA per tile (K=64). Output row stride PW=448.
// ---------------------------------------------------------------------------
__global__ __launch_bounds__(256)
void gemm_qkvs(const float* __restrict__ x,
               const _Float16* __restrict__ bpack,
               const float* __restrict__ bcat,
               float* __restrict__ P)
{
  __shared__ __align__(32) _Float16 ldsA[2 * 32 * 16];   // fragment-ordered A
  const int tid   = threadIdx.x;
  const int mbase = blockIdx.x * 16;

  // Stage x-tile directly in A-fragment order: element u of ldsA is
  // (s = u>>9, lane = (u>>4)&31, e = u&15) -> row = lane&15,
  // K = 32*s + (lane<16?0:16) + e.  Each thread: one float4 load.
  {
    int u0  = tid * 4;
    int s   = u0 >> 9;
    int ln  = (u0 >> 4) & 31;
    int e0  = u0 & 15;                        // multiple of 4 -> K contiguous
    int row = ln & 15;
    int K   = 32 * s + ((ln < 16) ? 0 : 16) + e0;
    const float4 xv = *(const float4*)(x + (size_t)(mbase + row) * DIN + K);
    v4h hv = {(_Float16)xv.x, (_Float16)xv.y, (_Float16)xv.z, (_Float16)xv.w};
    *(v4h*)&ldsA[u0] = hv;
  }
  __syncthreads();

  const int lane = tid & 31;
  const int wave = tid >> 5;

  const v16h a0 = *(const v16h*)&ldsA[(0 * 32 + lane) * 16];
  const v16h a1 = *(const v16h*)&ldsA[(1 * 32 + lane) * 16];

  const int rb = (lane < 16) ? 0 : 8;         // C/D layout: VGPR r -> row M=r(+8)

  for (int nt = wave; nt < NTILES; nt += 8) {
    const v16h b0 = *(const v16h*)(bpack + ((size_t)(nt * 2 + 0) * 32 + lane) * 16);
    const v16h b1 = *(const v16h*)(bpack + ((size_t)(nt * 2 + 1) * 32 + lane) * 16);

    v8f c = {};
    c = __builtin_amdgcn_wmma_f32_16x16x32_f16(false, a0, false, b0, (short)0, c, false, false);
    c = __builtin_amdgcn_wmma_f32_16x16x32_f16(false, a1, false, b1, (short)0, c, false, false);

    int   n    = nt * 16 + (lane & 15);       // grp*128 + col + (lane&15) == nt*16 + ...
    float bias = bcat[n];
    #pragma unroll
    for (int r = 0; r < 8; ++r) {
      P[(size_t)(mbase + rb + r) * PW + n] = c[r] + bias;
    }
  }
}

// ---------------------------------------------------------------------------
__global__ void init_layer(float* __restrict__ acc, float* __restrict__ amax,
                           float* __restrict__ denom)
{
  int t = blockIdx.x * 256 + threadIdx.x;
  if (t < NNODES * CH) acc[t] = 0.0f;
  if (t < NNODES * NH) {
    amax[t]  = __int_as_float(0xFF800000);    // -inf
    denom[t] = 0.0f;
  }
}

// One wave32 per edge: both heads' q.k dot products + running segment max.
__global__ __launch_bounds__(256)
void edge_alpha(const float* __restrict__ P, const int* __restrict__ eidx,
                float* __restrict__ alpha, float* __restrict__ amax)
{
  int e = (blockIdx.x * 256 + threadIdx.x) >> 5;
  if (e >= NEDGES) return;
  int lane = threadIdx.x & 31;
  int src  = eidx[e];
  int dst  = eidx[NEDGES + e];
  int head = lane >> 4;
  int c    = (lane & 15) * 4;

  const float4 qi = *(const float4*)(P + (size_t)dst * PW + head * CH + c);         // Q
  const float4 kj = *(const float4*)(P + (size_t)src * PW + 128 + head * CH + c);   // K
  float p = qi.x * kj.x + qi.y * kj.y + qi.z * kj.z + qi.w * kj.w;
  p += __shfl_xor(p, 1);
  p += __shfl_xor(p, 2);
  p += __shfl_xor(p, 4);
  p += __shfl_xor(p, 8);                      // reduce within each 16-lane head group

  if ((lane & 15) == 0) {
    float a = p * 0.125f;                     // 1/sqrt(C), C=64
    alpha[(size_t)e * NH + head] = a;
    atomicMaxF(&amax[(size_t)dst * NH + head], a);
  }
}

// One thread per (edge, head): exponentiate and accumulate denominator.
__global__ void edge_exp(const int* __restrict__ eidx, float* __restrict__ alpha,
                         const float* __restrict__ amax, float* __restrict__ denom)
{
  int t = blockIdx.x * 256 + threadIdx.x;
  if (t >= NEDGES * NH) return;
  int e = t >> 1, h = t & 1;
  int dst = eidx[NEDGES + e];
  float ex = __expf(alpha[t] - amax[dst * NH + h]);
  alpha[t] = ex;                              // reuse buffer for exp values
  atomicAdd(&denom[dst * NH + h], ex);
}

// One wave32 per edge: normalized weights, head-mean fused, scatter-add.
__global__ __launch_bounds__(256)
void edge_agg(const float* __restrict__ P, const int* __restrict__ eidx,
              const float* __restrict__ exw, const float* __restrict__ denom,
              float* __restrict__ acc)
{
  int e = (blockIdx.x * 256 + threadIdx.x) >> 5;
  if (e >= NEDGES) return;
  int lane = threadIdx.x & 31;
  int src  = eidx[e];
  int dst  = eidx[NEDGES + e];

  float w0 = exw[(size_t)e * NH + 0] / (denom[dst * NH + 0] + 1e-16f);
  float w1 = exw[(size_t)e * NH + 1] / (denom[dst * NH + 1] + 1e-16f);

  int c = lane * 2;
  const float2 v0 = *(const float2*)(P + (size_t)src * PW + 256 + c);        // V head 0
  const float2 v1 = *(const float2*)(P + (size_t)src * PW + 256 + CH + c);   // V head 1
  float r0 = 0.5f * (w0 * v0.x + w1 * v1.x);
  float r1 = 0.5f * (w0 * v0.y + w1 * v1.y);
  atomicAdd(&acc[(size_t)dst * CH + c + 0], r0);
  atomicAdd(&acc[(size_t)dst * CH + c + 1], r1);
}

// out = aggregated mean + skip projection (bias already folded into P's S).
__global__ void finalize_layer(const float* __restrict__ acc,
                               const float* __restrict__ P,
                               float* __restrict__ out)
{
  int t = blockIdx.x * 256 + threadIdx.x;
  if (t < NNODES * CH) {
    int node = t >> 6, c = t & 63;
    out[t] = acc[t] + P[(size_t)node * PW + 384 + c];
  }
}

// ---------------------------------------------------------------------------
static void run_layer(const float* x, const float* const* W, const int* eidx,
                      float* P, float* alpha, float* amax, float* denom,
                      float* acc, float* bcat, _Float16* bpack,
                      float* out, hipStream_t stream)
{
  pack_weights<<<(BPACKN + 255) / 256, 256, 0, stream>>>(
      W[0], W[1], W[2], W[3], W[4], W[5], W[6], W[7], bpack, bcat);
  gemm_qkvs<<<NNODES / 16, 256, 0, stream>>>(x, bpack, bcat, P);
  init_layer<<<(NNODES * CH + 255) / 256, 256, 0, stream>>>(acc, amax, denom);
  edge_alpha<<<(NEDGES * 32 + 255) / 256, 256, 0, stream>>>(P, eidx, alpha, amax);
  edge_exp<<<(NEDGES * NH + 255) / 256, 256, 0, stream>>>(eidx, alpha, amax, denom);
  edge_agg<<<(NEDGES * 32 + 255) / 256, 256, 0, stream>>>(P, eidx, alpha, denom, acc);
  finalize_layer<<<(NNODES * CH + 255) / 256, 256, 0, stream>>>(acc, P, out);
}

extern "C" void kernel_launch(void* const* d_in, const int* in_sizes, int n_in,
                              void* d_out, int out_size, void* d_ws, size_t ws_size,
                              hipStream_t stream)
{
  (void)in_sizes; (void)n_in; (void)out_size; (void)ws_size;
  const float* x    = (const float*)d_in[0];
  const int*   eidx = (const int*)d_in[1];

  const float* W1[8];
  const float* W2[8];
  for (int i = 0; i < 8; ++i) {
    W1[i] = (const float*)d_in[2 + i];
    W2[i] = (const float*)d_in[10 + i];
  }

  float* ws    = (float*)d_ws;
  float* P     = ws;                              // N*448  (Q|K|V|S per node)
  float* alpha = P + (size_t)NNODES * PW;         // E*2
  float* amax  = alpha + (size_t)NEDGES * NH;     // N*2
  float* denom = amax + (size_t)NNODES * NH;      // N*2
  float* acc   = denom + (size_t)NNODES * NH;     // N*64 (also layer-1 output h1)
  float* bcat  = acc + (size_t)NNODES * CH;       // 448
  _Float16* bpack = (_Float16*)(bcat + PW);       // 28672 halves (32B aligned)

  // Layer 1: output written in-place into acc (acc += skip) -> h1.
  run_layer(x, W1, eidx, P, alpha, amax, denom, acc, bcat, bpack, acc, stream);
  // Layer 2: reads h1 (=acc) in its GEMM (stream-ordered before acc is
  // re-zeroed by init_layer), writes final result to d_out.
  run_layer(acc, W2, eidx, P, alpha, amax, denom, acc, bcat, bpack,
            (float*)d_out, stream);
}
